// LlamaDecoderLayer_83236466197159
// MI455X (gfx1250) — compile-verified, matrix operand order checked by reading, not run on hardware
//
#include <hip/hip_runtime.h>
#include <hip/hip_bf16.h>
#include <math.h>

// ---------------------------------------------------------------------------
// Types for CDNA5 WMMA (wave32): v_wmma_f32_16x16x32_bf16
// ---------------------------------------------------------------------------
typedef __attribute__((ext_vector_type(16))) __bf16 v16bf;
typedef __attribute__((ext_vector_type(8)))  __bf16 v8bf;
typedef __attribute__((ext_vector_type(8)))  float  v8f;
typedef __attribute__((ext_vector_type(4)))  int    v4i;

union V16U { v16bf v; v8bf h[2]; };

static __device__ __forceinline__ v8f wmma_bf16(v16bf a, v16bf b, v8f c) {
  // 8 args: (neg_a, A, neg_b, B, c_mod, C, reuse_a, reuse_b)
  return __builtin_amdgcn_wmma_f32_16x16x32_bf16(false, a, false, b, (short)0, c,
                                                 false, false);
}

static __device__ __forceinline__ v8f v8f_zero() {
  v8f z = {0.f, 0.f, 0.f, 0.f, 0.f, 0.f, 0.f, 0.f};
  return z;
}

// ---------------------------------------------------------------------------
// CDNA5 async global->LDS copy (ASYNCcnt path), with safe fallback
// Probe round 2 showed the builtin exists with signature:
//   (int4 addrspace(1)*, int4 addrspace(3)*, imm offset, imm cpol)
// ---------------------------------------------------------------------------
#if defined(__HIP_DEVICE_COMPILE__) && \
    __has_builtin(__builtin_amdgcn_global_load_async_to_lds_b128)
#define HAVE_ASYNC_LDS 1
#else
#define HAVE_ASYNC_LDS 0
#endif

static __device__ __forceinline__ void copy16_g2l(const void* g, void* l) {
#if HAVE_ASYNC_LDS
  __builtin_amdgcn_global_load_async_to_lds_b128(
      (__attribute__((address_space(1))) v4i*)const_cast<void*>(g),
      (__attribute__((address_space(3))) v4i*)l, 0, 0);
#else
  *(uint4*)l = *(const uint4*)g;
#endif
}

static __device__ __forceinline__ void async_wait_all() {
#if HAVE_ASYNC_LDS
#if __has_builtin(__builtin_amdgcn_s_wait_asynccnt)
  __builtin_amdgcn_s_wait_asynccnt(0);
#else
  asm volatile("s_wait_asynccnt 0x0" ::: "memory");
#endif
#endif
}

// Problem constants (from the reference)
constexpr int Bn = 2, Sn = 1024, Hn = 4096, NQn = 32, NKVn = 8, Dn = 128, In = 11008;
constexpr int TOK = Bn * Sn;                 // 2048 tokens

// ---------------------------------------------------------------------------
// Transposing f32->bf16 weight conversion: in[K][N] -> out[N][K]
// 64x64 tiles through LDS; both global sides coalesced.
// ---------------------------------------------------------------------------
__global__ __launch_bounds__(256)
void k_cvt_bf16_t(const float* __restrict__ in, __bf16* __restrict__ out,
                  int K, int N) {
  __shared__ __bf16 tile[64][65];
  const int tid = threadIdx.x;
  const int tx = tid & 63, ty = tid >> 6;         // 64 x 4 threads
  const int k0 = blockIdx.y * 64, n0 = blockIdx.x * 64;
#pragma unroll
  for (int r = 0; r < 16; ++r) {
    const int row = r * 4 + ty;                   // 0..63
    tile[row][tx] = (__bf16)in[(size_t)(k0 + row) * N + (n0 + tx)];
  }
  __syncthreads();
#pragma unroll
  for (int r = 0; r < 16; ++r) {
    const int row = r * 4 + ty;                   // output row (n)
    out[(size_t)(n0 + row) * K + (k0 + tx)] = tile[tx][row];
  }
}

// ---------------------------------------------------------------------------
// RMSNorm: f32 in, bf16 out (feeds the next WMMA GEMM)
// ---------------------------------------------------------------------------
__global__ __launch_bounds__(256)
void k_rmsnorm_bf16(const float* __restrict__ x, const float* __restrict__ wgt,
                    __bf16* __restrict__ out) {
  __shared__ float red[256];
  const int row = blockIdx.x, tid = threadIdx.x;
  const float* xr = x + (size_t)row * Hn;
  float ss = 0.f;
  for (int i = tid; i < Hn / 4; i += 256) {
    float4 v = ((const float4*)xr)[i];
    ss += v.x * v.x + v.y * v.y + v.z * v.z + v.w * v.w;
  }
  red[tid] = ss;
  __syncthreads();
  for (int s = 128; s > 0; s >>= 1) {
    if (tid < s) red[tid] += red[tid + s];
    __syncthreads();
  }
  const float r = rsqrtf(red[0] / (float)Hn + 1e-5f);
  for (int i = tid; i < Hn; i += 256)
    out[(size_t)row * Hn + i] = (__bf16)(xr[i] * r * wgt[i]);
}

// ---------------------------------------------------------------------------
// bf16 WMMA GEMM: C[M,N](f32) = A[M,K](bf16, row-major) x B^T[N,K](bf16) (+R)
// Block tile 128x128, 8 waves (2x4), wave tile 64x32 (4x2 WMMA tiles), K-step 32.
// Both tiles staged with 16B (async) copies; no scalar LDS traffic.
// ---------------------------------------------------------------------------
template <int ADD_RES>
__global__ __launch_bounds__(256)
void k_gemm_bf16(const __bf16* __restrict__ A, const __bf16* __restrict__ BT,
                 const float* __restrict__ R, float* __restrict__ C,
                 int M, int N, int K) {
  __shared__ __align__(16) __bf16 As[128][40];   // rows padded to 40 bf16 (80B)
  __shared__ __align__(16) __bf16 Bs[128][40];   // B^T tile [n][k]
  const int tid = threadIdx.x;
  const int lane = tid & 31, wid = tid >> 5;
  const int wr = wid & 1, wc = wid >> 1;         // 2x4 wave grid
  const int lh = lane >> 4, mrel = lane & 15;    // lane half, lane-in-half
  const int rowBase = blockIdx.y * 128, colBase = blockIdx.x * 128;

  v8f acc[4][2];
#pragma unroll
  for (int i = 0; i < 4; ++i)
#pragma unroll
    for (int j = 0; j < 2; ++j) acc[i][j] = v8f_zero();

  const int ldRow = tid >> 1, ldK = (tid & 1) * 16;   // 2 threads per tile row

  for (int k0 = 0; k0 < K; k0 += 32) {
    const __bf16* ag = A  + (size_t)(rowBase + ldRow) * K + (k0 + ldK);
    const __bf16* bg = BT + (size_t)(colBase + ldRow) * K + (k0 + ldK);
    copy16_g2l(ag,     &As[ldRow][ldK]);
    copy16_g2l(ag + 8, &As[ldRow][ldK + 8]);
    copy16_g2l(bg,     &Bs[ldRow][ldK]);
    copy16_g2l(bg + 8, &Bs[ldRow][ldK + 8]);
    if (k0 + 32 < K) {  // emits global_prefetch_b8 for the next K-tile
      __builtin_prefetch(ag + 32, 0, 0);
      __builtin_prefetch(bg + 32, 0, 0);
    }
    async_wait_all();
    __syncthreads();

    // Fragments per ISA 7.12.2: lanes<16 K={0..7,16..23}, lanes>=16 K={8..15,24..31}
    v16bf af[4], bfv[2];
#pragma unroll
    for (int rs = 0; rs < 4; ++rs) {
      V16U u;
      const int m = wr * 64 + rs * 16 + mrel;
      u.h[0] = *(const v8bf*)(&As[m][lh * 8]);
      u.h[1] = *(const v8bf*)(&As[m][lh * 8 + 16]);
      af[rs] = u.v;
    }
#pragma unroll
    for (int cs = 0; cs < 2; ++cs) {
      V16U u;
      const int n = wc * 32 + cs * 16 + mrel;
      u.h[0] = *(const v8bf*)(&Bs[n][lh * 8]);
      u.h[1] = *(const v8bf*)(&Bs[n][lh * 8 + 16]);
      bfv[cs] = u.v;
    }
#pragma unroll
    for (int rs = 0; rs < 4; ++rs)
#pragma unroll
      for (int cs = 0; cs < 2; ++cs)
        acc[rs][cs] = wmma_bf16(af[rs], bfv[cs], acc[rs][cs]);
    __syncthreads();
  }

  // Epilogue: C layout lane=N, VGPR j = M row (+8 for upper half)
#pragma unroll
  for (int rs = 0; rs < 4; ++rs)
#pragma unroll
    for (int cs = 0; cs < 2; ++cs)
#pragma unroll
      for (int j = 0; j < 8; ++j) {
        const int r = rowBase + wr * 64 + rs * 16 + lh * 8 + j;
        const int c = colBase + wc * 32 + cs * 16 + mrel;
        float v = acc[rs][cs][j];
        if (ADD_RES) v += R[(size_t)r * N + c];
        C[(size_t)r * N + c] = v;
      }
}

// ---------------------------------------------------------------------------
// RoPE + permute: f32 [B*S, NH*D] -> bf16 [B][NH][S][D]
// ---------------------------------------------------------------------------
__global__ __launch_bounds__(Dn)
void k_rope_bf16(const float* __restrict__ in, const float* __restrict__ cosb,
                 const float* __restrict__ sinb, __bf16* __restrict__ out, int NH) {
  const int d = threadIdx.x, h = blockIdx.x, s = blockIdx.y, b = blockIdx.z;
  const size_t irow = ((size_t)(b * Sn + s) * NH + h) * Dn;
  const float v  = in[irow + d];
  const float vr = (d < Dn / 2) ? -in[irow + d + Dn / 2] : in[irow + d - Dn / 2];
  out[(((size_t)b * NH + h) * Sn + s) * Dn + d] =
      (__bf16)(v * cosb[s * Dn + d] + vr * sinb[s * Dn + d]);
}

// V permute to d-major: f32 [B*S, NKV*D] -> bf16 [B][NKV][D][S]
__global__ __launch_bounds__(256)
void k_permute_vT(const float* __restrict__ in, __bf16* __restrict__ out) {
  const int h = blockIdx.x, d = blockIdx.y, b = blockIdx.z;
  for (int s = threadIdx.x; s < Sn; s += 256)
    out[(((size_t)b * NKVn + h) * Dn + d) * Sn + s] =
        (__bf16)in[((size_t)(b * Sn + s) * NKVn + h) * Dn + d];
}

// ---------------------------------------------------------------------------
// Flash attention (causal, GQA 4:1). Block = 4 waves, 64 q rows per block.
// Each wave: 16 q rows, online softmax, QK^T and P*V via WMMA bf16.
// K staged row-major; V pre-transposed in global, staged with 16B copies.
// ---------------------------------------------------------------------------
__global__ __launch_bounds__(128)
void k_flash_attn(const __bf16* __restrict__ Q, const __bf16* __restrict__ Kg,
                  const __bf16* __restrict__ VTg, __bf16* __restrict__ O,
                  float scale) {
  __shared__ __align__(16) __bf16 Ks[32][136];    // K block [key][d], padded
  __shared__ __align__(16) __bf16 Vt[128][40];    // V block [d][key], padded
  __shared__ __align__(16) __bf16 Ps[4][16][40];  // per-wave P tile [m][key]
  const int tid = threadIdx.x;
  const int lane = tid & 31, w = tid >> 5;
  const int lh = lane >> 4, mrel = lane & 15;
  const int qblk = blockIdx.x, h = blockIdx.y, b = blockIdx.z;
  const int kvh = h >> 2;                         // NQ/NKV = 4
  const int q0w = qblk * 64 + w * 16;

  // Preload Q fragments for all 4 d-chunks of 32
  v16bf qf[4];
  {
    const __bf16* qg = Q + (((size_t)(b * NQn + h)) * Sn + (q0w + mrel)) * Dn;
#pragma unroll
    for (int c = 0; c < 4; ++c) {
      V16U u;
      u.h[0] = *(const v8bf*)(qg + c * 32 + lh * 8);
      u.h[1] = *(const v8bf*)(qg + c * 32 + lh * 8 + 16);
      qf[c] = u.v;
    }
  }

  v8f acc[8];
#pragma unroll
  for (int t = 0; t < 8; ++t) acc[t] = v8f_zero();
  float mrow[8], lrow[8];
#pragma unroll
  for (int j = 0; j < 8; ++j) { mrow[j] = -__builtin_inff(); lrow[j] = 0.f; }

  const __bf16* kb_base = Kg  + ((size_t)(b * NKVn + kvh)) * Sn * Dn;  // [key][d]
  const __bf16* vt_base = VTg + ((size_t)(b * NKVn + kvh)) * Dn * Sn;  // [d][key]
  const int nkb = qblk * 2 + 2;                   // uniform per block (causal)

  for (int kb = 0; kb < nkb; ++kb) {
    const int key0 = kb * 32;
    __syncthreads();  // protect Ks/Vt reuse across iterations
    {  // K block: 32 keys x 128 d, 4 x 16B per thread
      const int key = tid >> 2, dq = (tid & 3) * 32;
      const __bf16* src = kb_base + (size_t)(key0 + key) * Dn + dq;
#pragma unroll
      for (int i = 0; i < 4; ++i)
        copy16_g2l(src + 8 * i, &Ks[key][dq + 8 * i]);
    }
    {  // V block [d][key]: thread = d row, 32 keys contiguous, 4 x 16B
      const __bf16* src = vt_base + (size_t)tid * Sn + key0;
#pragma unroll
      for (int i = 0; i < 4; ++i)
        copy16_g2l(src + 8 * i, &Vt[tid][8 * i]);
    }
    async_wait_all();
    __syncthreads();

    // scores S = Q K^T over d=128 (4 WMMA per 16x16 tile, 2 key tiles)
    v8f s0 = v8f_zero(), s1 = v8f_zero();
#pragma unroll
    for (int c = 0; c < 4; ++c) {
      V16U u0, u1;
      u0.h[0] = *(const v8bf*)(&Ks[mrel][c * 32 + lh * 8]);
      u0.h[1] = *(const v8bf*)(&Ks[mrel][c * 32 + lh * 8 + 16]);
      u1.h[0] = *(const v8bf*)(&Ks[16 + mrel][c * 32 + lh * 8]);
      u1.h[1] = *(const v8bf*)(&Ks[16 + mrel][c * 32 + lh * 8 + 16]);
      s0 = wmma_bf16(qf[c], u0.v, s0);
      s1 = wmma_bf16(qf[c], u1.v, s1);
    }

    // Online softmax, per row j (row = q0w + lh*8 + j, col = key0 + {0,16} + mrel)
#pragma unroll
    for (int j = 0; j < 8; ++j) {
      const int qrow = q0w + lh * 8 + j;
      float a0 = s0[j] * scale;
      float a1 = s1[j] * scale;
      if (key0 + mrel > qrow)      a0 = -__builtin_inff();
      if (key0 + 16 + mrel > qrow) a1 = -__builtin_inff();
      float mloc = fmaxf(a0, a1);
#pragma unroll
      for (int msk = 1; msk < 16; msk <<= 1)
        mloc = fmaxf(mloc, __shfl_xor(mloc, msk, 32));
      const float mnew  = fmaxf(mrow[j], mloc);
      const float alpha = __expf(mrow[j] - mnew);
      const float p0 = __expf(a0 - mnew);
      const float p1 = __expf(a1 - mnew);
      float psum = p0 + p1;
#pragma unroll
      for (int msk = 1; msk < 16; msk <<= 1) psum += __shfl_xor(psum, msk, 32);
      lrow[j] = lrow[j] * alpha + psum;
      mrow[j] = mnew;
#pragma unroll
      for (int t = 0; t < 8; ++t) acc[t][j] *= alpha;
      Ps[w][lh * 8 + j][mrel]      = (__bf16)p0;
      Ps[w][lh * 8 + j][16 + mrel] = (__bf16)p1;
    }
    __syncthreads();  // make Ps visible for A-fragment reload

    // O += P V : A = P[16x32], B = V^T fragments from Vt
    V16U pu;
    pu.h[0] = *(const v8bf*)(&Ps[w][mrel][lh * 8]);
    pu.h[1] = *(const v8bf*)(&Ps[w][mrel][lh * 8 + 16]);
#pragma unroll
    for (int t = 0; t < 8; ++t) {
      V16U vu;
      vu.h[0] = *(const v8bf*)(&Vt[t * 16 + mrel][lh * 8]);
      vu.h[1] = *(const v8bf*)(&Vt[t * 16 + mrel][lh * 8 + 16]);
      acc[t] = wmma_bf16(pu.v, vu.v, acc[t]);
    }
  }

  // Epilogue: out token-major bf16 [B*S][NQ*D]
#pragma unroll
  for (int j = 0; j < 8; ++j) {
    const int qrow = q0w + lh * 8 + j;
    const float inv = 1.0f / lrow[j];
    __bf16* orow = O + (size_t)(b * Sn + qrow) * (NQn * Dn) + h * Dn;
#pragma unroll
    for (int t = 0; t < 8; ++t)
      orow[t * 16 + mrel] = (__bf16)(acc[t][j] * inv);
  }
}

// ---------------------------------------------------------------------------
// silu(g) * u -> bf16
// ---------------------------------------------------------------------------
__global__ __launch_bounds__(256)
void k_silu_mul(const float* __restrict__ g, const float* __restrict__ u,
                __bf16* __restrict__ out, size_t n) {
  size_t i = (size_t)blockIdx.x * blockDim.x + threadIdx.x;
  size_t stride = (size_t)gridDim.x * blockDim.x;
  for (; i < n; i += stride) {
    const float gv = g[i];
    out[i] = (__bf16)(gv / (1.0f + __expf(-gv)) * u[i]);
  }
}

// ---------------------------------------------------------------------------
// Host orchestration
// ---------------------------------------------------------------------------
extern "C" void kernel_launch(void* const* d_in, const int* in_sizes, int n_in,
                              void* d_out, int out_size, void* d_ws, size_t ws_size,
                              hipStream_t stream) {
  const float* x    = (const float*)d_in[0];
  const float* cosb = (const float*)d_in[1];
  const float* sinb = (const float*)d_in[2];
  const float* anw  = (const float*)d_in[3];
  const float* fnw  = (const float*)d_in[4];
  const float* wq   = (const float*)d_in[5];
  const float* wk   = (const float*)d_in[6];
  const float* wv   = (const float*)d_in[7];
  const float* wo   = (const float*)d_in[8];
  const float* wg   = (const float*)d_in[9];
  const float* wu   = (const float*)d_in[10];
  const float* wd   = (const float*)d_in[11];
  float* out = (float*)d_out;

  char* p = (char*)d_ws;
  auto alloc = [&](size_t bytes) -> void* {
    void* r = (void*)p;
    p += (bytes + 255) & ~(size_t)255;
    return r;
  };
  const size_t nQW = (size_t)Hn * NQn * Dn;      // 4096*4096
  const size_t nKW = (size_t)Hn * NKVn * Dn;     // 4096*1024
  const size_t nGW = (size_t)Hn * In;            // 4096*11008

  __bf16* wqb  = (__bf16*)alloc(nQW * 2);        // all weights bf16, transposed [N][K]
  __bf16* wkb  = (__bf16*)alloc(nKW * 2);
  __bf16* wvb  = (__bf16*)alloc(nKW * 2);
  __bf16* wob  = (__bf16*)alloc(nQW * 2);
  __bf16* wgb  = (__bf16*)alloc(nGW * 2);
  __bf16* wub  = (__bf16*)alloc(nGW * 2);
  __bf16* wdb  = (__bf16*)alloc(nGW * 2);
  __bf16* hinb = (__bf16*)alloc((size_t)TOK * Hn * 2);
  float*  qf32 = (float*)alloc((size_t)TOK * NQn * Dn * 4);
  float*  kf32 = (float*)alloc((size_t)TOK * NKVn * Dn * 4);
  float*  vf32 = (float*)alloc((size_t)TOK * NKVn * Dn * 4);
  __bf16* Qb   = (__bf16*)alloc((size_t)TOK * NQn * Dn * 2);
  __bf16* Kb   = (__bf16*)alloc((size_t)TOK * NKVn * Dn * 2);
  __bf16* VTb  = (__bf16*)alloc((size_t)TOK * NKVn * Dn * 2);
  __bf16* attnb= (__bf16*)alloc((size_t)TOK * Hn * 2);
  float*  hres = (float*)alloc((size_t)TOK * Hn * 4);
  __bf16* minb = (__bf16*)alloc((size_t)TOK * Hn * 2);
  float*  gf   = (float*)alloc((size_t)TOK * In * 4);
  float*  uf   = (float*)alloc((size_t)TOK * In * 4);
  __bf16* gub  = (__bf16*)alloc((size_t)TOK * In * 2);

  // 1) weights -> bf16, transposed to [N][K] (GEMM B-operand layout)
  k_cvt_bf16_t<<<dim3((NQn * Dn) / 64, Hn / 64), 256, 0, stream>>>(wq, wqb, Hn, NQn * Dn);
  k_cvt_bf16_t<<<dim3((NKVn * Dn) / 64, Hn / 64), 256, 0, stream>>>(wk, wkb, Hn, NKVn * Dn);
  k_cvt_bf16_t<<<dim3((NKVn * Dn) / 64, Hn / 64), 256, 0, stream>>>(wv, wvb, Hn, NKVn * Dn);
  k_cvt_bf16_t<<<dim3(Hn / 64, (NQn * Dn) / 64), 256, 0, stream>>>(wo, wob, NQn * Dn, Hn);
  k_cvt_bf16_t<<<dim3(In / 64, Hn / 64), 256, 0, stream>>>(wg, wgb, Hn, In);
  k_cvt_bf16_t<<<dim3(In / 64, Hn / 64), 256, 0, stream>>>(wu, wub, Hn, In);
  k_cvt_bf16_t<<<dim3(Hn / 64, In / 64), 256, 0, stream>>>(wd, wdb, In, Hn);

  // 2) attn RMSNorm -> bf16
  k_rmsnorm_bf16<<<TOK, 256, 0, stream>>>(x, anw, hinb);

  // 3) QKV projections (WMMA)
  k_gemm_bf16<0><<<dim3((NQn * Dn) / 128, TOK / 128), 256, 0, stream>>>(
      hinb, wqb, nullptr, qf32, TOK, NQn * Dn, Hn);
  k_gemm_bf16<0><<<dim3((NKVn * Dn) / 128, TOK / 128), 256, 0, stream>>>(
      hinb, wkb, nullptr, kf32, TOK, NKVn * Dn, Hn);
  k_gemm_bf16<0><<<dim3((NKVn * Dn) / 128, TOK / 128), 256, 0, stream>>>(
      hinb, wvb, nullptr, vf32, TOK, NKVn * Dn, Hn);

  // 4) RoPE + layout permutes
  k_rope_bf16<<<dim3(NQn, Sn, Bn), Dn, 0, stream>>>(qf32, cosb, sinb, Qb, NQn);
  k_rope_bf16<<<dim3(NKVn, Sn, Bn), Dn, 0, stream>>>(kf32, cosb, sinb, Kb, NKVn);
  k_permute_vT<<<dim3(NKVn, Dn, Bn), 256, 0, stream>>>(vf32, VTb);

  // 5) causal flash attention (WMMA)
  k_flash_attn<<<dim3(Sn / 64, NQn, Bn), 128, 0, stream>>>(
      Qb, Kb, VTb, attnb, 1.0f / sqrtf((float)Dn));

  // 6) output projection + residual: h = x + attn @ wo
  k_gemm_bf16<1><<<dim3(Hn / 128, TOK / 128), 256, 0, stream>>>(
      attnb, wob, x, hres, TOK, Hn, NQn * Dn);

  // 7) ffn RMSNorm -> bf16
  k_rmsnorm_bf16<<<TOK, 256, 0, stream>>>(hres, fnw, minb);

  // 8) gate / up (WMMA)
  k_gemm_bf16<0><<<dim3(In / 128, TOK / 128), 256, 0, stream>>>(
      minb, wgb, nullptr, gf, TOK, In, Hn);
  k_gemm_bf16<0><<<dim3(In / 128, TOK / 128), 256, 0, stream>>>(
      minb, wub, nullptr, uf, TOK, In, Hn);

  // 9) silu(g)*u -> bf16
  k_silu_mul<<<4096, 256, 0, stream>>>(gf, uf, gub, (size_t)TOK * In);

  // 10) down projection + residual -> d_out (f32)
  k_gemm_bf16<1><<<dim3(Hn / 128, TOK / 128), 256, 0, stream>>>(
      gub, wdb, hres, out, TOK, Hn, In);
}
